// GraphConv_67585605370302
// MI455X (gfx1250) — matrix operand order repaired and checked by reading, not compiled, not run on hardware
//
#include <hip/hip_runtime.h>
#include <hip/hip_bf16.h>

// ---------- types for WMMA ----------
typedef __bf16 v16bf __attribute__((ext_vector_type(16)));
typedef float  v8f   __attribute__((ext_vector_type(8)));
typedef unsigned int u32x4 __attribute__((ext_vector_type(4)));

union FragAB {                 // one wave-striped A or B operand (16 bf16 / lane)
    v16bf v;
    unsigned short s[16];
    u32x4 q[2];
};

// ---------- bf16 helpers (manual, RNE) ----------
__device__ __forceinline__ unsigned short f2bf(float f) {
    unsigned u = __float_as_uint(f);
    u += 0x7FFFu + ((u >> 16) & 1u);
    return (unsigned short)(u >> 16);
}
__device__ __forceinline__ float bf2f(unsigned short h) {
    return __uint_as_float(((unsigned)h) << 16);
}

// ---------- ordered-uint encoding for float atomicMax ----------
__device__ __forceinline__ unsigned f2ord(float f) {
    unsigned u = __float_as_uint(f);
    return (u & 0x80000000u) ? ~u : (u | 0x80000000u);
}
__device__ __forceinline__ float ord2f(unsigned o) {
    return (o & 0x80000000u) ? __uint_as_float(o ^ 0x80000000u)
                             : __uint_as_float(~o);
}

// ---------- tiny utility kernels ----------
__global__ void zero_kernel(float* __restrict__ p, size_t n) {
    size_t i = (size_t)blockIdx.x * blockDim.x + threadIdx.x;
    if (i < n) p[i] = 0.0f;
}

__global__ void f32_to_bf16_kernel(const float* __restrict__ src,
                                   unsigned short* __restrict__ dst, int n) {
    int i = blockIdx.x * blockDim.x + threadIdx.x;
    if (i < n) dst[i] = f2bf(src[i]);
}

__global__ void degree_kernel(const int* __restrict__ col,
                              const float* __restrict__ ew,
                              float* __restrict__ deg, int E) {
    int e = blockIdx.x * blockDim.x + threadIdx.x;
    if (e < E) atomicAdd(&deg[col[e]], ew[e]);
}

// deg -> dinv in place; self-loop contributes exactly +1 to every node
__global__ void dinv_kernel(float* __restrict__ deg, int N) {
    int i = blockIdx.x * blockDim.x + threadIdx.x;
    if (i < N) deg[i] = rsqrtf(deg[i] + 1.0f);
}

// ---------- GEMM1: t1[N,128] = bf16( embed[x_ids] @ W1 ), via WMMA bf16 ----------
__global__ void __launch_bounds__(256)
gemm1_kernel(const int* __restrict__ xids,
             const unsigned short* __restrict__ emb,   // [VOCAB][64] bf16
             const unsigned short* __restrict__ w1,    // [64][128]  bf16
             unsigned short* __restrict__ t1,          // [N][128]   bf16
             int N, int tilesPerBlock)
{
    const int lane  = threadIdx.x & 31;
    const int wave  = threadIdx.x >> 5;     // 8 waves -> 8 N-tiles of 16 (128 cols)
    const int n0    = wave * 16;
    const int nlane = lane & 15;
    const int hi    = lane >> 4;            // 0: lanes 0-15, 1: lanes 16-31
    const int khalf = hi * 16;              // unused; B is loaded with koff like A
    const int koff  = hi * 8;               // A: K sub-offset per lane half

    // Load B fragments once per wave (kb = 0, 32); reused for all M tiles.
    FragAB b0, b1;
#pragma unroll
    for (int e = 0; e < 16; ++e) {
        b0.s[e] = w1[(0  + koff + (e & 7) + (e >> 3) * 16) * 128 + n0 + nlane];
        b1.s[e] = w1[(32 + koff + (e & 7) + (e >> 3) * 16) * 128 + n0 + nlane];
    }

    const int mbase = blockIdx.x * (16 * tilesPerBlock);
    for (int t = 0; t < tilesPerBlock; ++t) {
        const int m0 = mbase + t * 16;
        if (m0 >= N) break;                 // uniform across the block
        int m = m0 + nlane; if (m >= N) m = N - 1;
        const unsigned short* ar = emb + (size_t)xids[m] * 64;

        FragAB a0, a1;                      // A layout per ISA 7.12.2 (16-bit 16x32)
        a0.q[0] = *(const u32x4*)(ar + 0  + koff);
        a0.q[1] = *(const u32x4*)(ar + 16 + koff);
        a1.q[0] = *(const u32x4*)(ar + 32 + koff);
        a1.q[1] = *(const u32x4*)(ar + 48 + koff);

        v8f acc = {0.f,0.f,0.f,0.f,0.f,0.f,0.f,0.f};
        acc = __builtin_amdgcn_wmma_f32_16x16x32_bf16(false, a0.v, false, b0.v,
                                                      (short)0, acc, false, false);
        acc = __builtin_amdgcn_wmma_f32_16x16x32_bf16(false, a1.v, false, b1.v,
                                                      (short)0, acc, false, false);

        // C/D layout: VGPR r -> M = r + 8*hi, N = nlane
        unsigned short* st = t1 + (size_t)(m0 + hi * 8) * 128 + n0 + nlane;
        if (m0 + 16 <= N) {                 // uniform: full tile, unguarded stores
#pragma unroll
            for (int r = 0; r < 8; ++r) st[(size_t)r * 128] = f2bf(acc[r]);
        } else {                            // partial tail tile
#pragma unroll
            for (int r = 0; r < 8; ++r)
                if (m0 + r + hi * 8 < N) st[(size_t)r * 128] = f2bf(acc[r]);
        }
    }
}

// ---------- GEMM2: t2[N,64] = bf16( h1 @ W2 ), K=128 ----------
__global__ void __launch_bounds__(256)
gemm2_kernel(const unsigned short* __restrict__ h1,   // [N][128] bf16
             const unsigned short* __restrict__ w2,   // [128][64] bf16
             unsigned short* __restrict__ t2,         // [N][64]  bf16
             int N, int tilesPerBlock)
{
    const int lane  = threadIdx.x & 31;
    const int wave  = threadIdx.x >> 5;     // 8 waves: 4 N-tiles x 2 M-subtiles
    const int n0    = (wave & 3) * 16;
    const int msub  = (wave >> 2) * 16;
    const int nlane = lane & 15;
    const int hi    = lane >> 4;
    const int khalf = hi * 16;
    const int koff  = hi * 8;

    FragAB bw[4];
#pragma unroll
    for (int kk = 0; kk < 4; ++kk)
#pragma unroll
        for (int e = 0; e < 16; ++e)
            bw[kk].s[e] = w2[(kk * 32 + koff + (e & 7) + (e >> 3) * 16) * 64 + n0 + nlane];

    const int mbase = blockIdx.x * (32 * tilesPerBlock);
    for (int t = 0; t < tilesPerBlock; ++t) {
        if (mbase + t * 32 >= N) break;     // uniform across the block
        const int m0 = mbase + t * 32 + msub;
        int m = m0 + nlane; if (m >= N) m = N - 1;
        const unsigned short* ar = h1 + (size_t)m * 128;

        v8f acc = {0.f,0.f,0.f,0.f,0.f,0.f,0.f,0.f};
#pragma unroll
        for (int kk = 0; kk < 4; ++kk) {
            FragAB a;
            a.q[0] = *(const u32x4*)(ar + kk * 32 + koff);
            a.q[1] = *(const u32x4*)(ar + kk * 32 + 16 + koff);
            acc = __builtin_amdgcn_wmma_f32_16x16x32_bf16(false, a.v, false, bw[kk].v,
                                                          (short)0, acc, false, false);
        }

        unsigned short* st = t2 + (size_t)(m0 + hi * 8) * 64 + n0 + nlane;
        if (m0 + 16 <= N) {                 // uniform per wave: unguarded stores
#pragma unroll
            for (int r = 0; r < 8; ++r) st[(size_t)r * 64] = f2bf(acc[r]);
        } else {
#pragma unroll
            for (int r = 0; r < 8; ++r)
                if (m0 + r + hi * 8 < N) st[(size_t)r * 64] = f2bf(acc[r]);
        }
    }
}

// ---------- edge gather/scatter: agg[col] += t[row] * norm ----------
// One wave (32 lanes) per edge; F/32 features per lane. Self-loop edges
// (row=col=i, w=1) are synthesized for edge ids in [E, E+N).
template <int F>
__global__ void scatter_kernel(const int* __restrict__ row,
                               const int* __restrict__ col,
                               const float* __restrict__ ew,
                               const float* __restrict__ dinv,
                               const unsigned short* __restrict__ t,
                               float* __restrict__ agg,
                               int E, int N)
{
    constexpr int FPL = F / 32;
    long long gid = (long long)blockIdx.x * blockDim.x + threadIdx.x;
    int edge = (int)(gid >> 5);
    int lane = (int)(gid & 31);
    if (edge >= E + N) return;

    int r, c; float wv;
    if (edge < E) { r = row[edge]; c = col[edge]; wv = ew[edge]; }
    else          { r = c = edge - E;             wv = 1.0f;     }

    const float norm = dinv[r] * wv * dinv[c];
    const unsigned short* tr = t + (size_t)r * F + lane * FPL;
    float* ac = agg + (size_t)c * F + lane * FPL;
#pragma unroll
    for (int j = 0; j < FPL; ++j)
        atomicAdd(&ac[j], bf2f(tr[j]) * norm);
}

// ---------- h1 = bf16(relu(agg1 + b1)) ----------
__global__ void bias_relu_kernel(const float* __restrict__ agg1,
                                 const float* __restrict__ b1,
                                 unsigned short* __restrict__ h1, int N)
{
    int i = blockIdx.x * blockDim.x + threadIdx.x;
    if (i >= N * 128) return;
    float v = agg1[i] + b1[i & 127];
    h1[i] = f2bf(v > 0.0f ? v : 0.0f);
}

// ---------- segment max pool (ordered-uint atomicMax) ----------
__global__ void pool_init_kernel(unsigned* __restrict__ outu, int n) {
    int i = blockIdx.x * blockDim.x + threadIdx.x;
    if (i < n) outu[i] = 0x007FFFFFu;       // f2ord(-inf)
}

__global__ void pool_max_kernel(const int* __restrict__ batch,
                                const float* __restrict__ agg2,
                                const float* __restrict__ b2,
                                unsigned* __restrict__ outu, int N)
{
    int gid = blockIdx.x * blockDim.x + threadIdx.x;
    if (gid >= N * 64) return;
    int n = gid >> 6, f = gid & 63;
    float v = agg2[(size_t)n * 64 + f] + b2[f];
    atomicMax(&outu[(size_t)batch[n] * 64 + f], f2ord(v));
}

__global__ void pool_fin_kernel(unsigned* __restrict__ outu, int n) {
    int i = blockIdx.x * blockDim.x + threadIdx.x;
    if (i < n) {
        float v = ord2f(outu[i]);
        ((float*)outu)[i] = v;
    }
}

// =====================================================================
extern "C" void kernel_launch(void* const* d_in, const int* in_sizes, int n_in,
                              void* d_out, int out_size, void* d_ws, size_t ws_size,
                              hipStream_t stream)
{
    const int*   x_ids = (const int*)  d_in[0];
    const int*   ei    = (const int*)  d_in[1];   // [2][E]
    const float* eattr = (const float*)d_in[2];
    const int*   batch = (const int*)  d_in[3];
    const float* embt  = (const float*)d_in[5];
    const float* W1    = (const float*)d_in[6];
    const float* b1    = (const float*)d_in[7];
    const float* W2    = (const float*)d_in[8];
    const float* b2    = (const float*)d_in[9];

    const int N = in_sizes[0];
    const int E = in_sizes[2];
    const int EMB = in_sizes[5];                  // VOCAB*64
    const int G = out_size / 64;

    const int* row = ei;
    const int* col = ei + E;

    // ---- workspace carve (256B aligned) ----
    auto al = [](size_t x) { return (x + 255) & ~(size_t)255; };
    char* p = (char*)d_ws;
    float*          deg  = (float*)p;          p += al((size_t)N * 4);
    unsigned short* w1b  = (unsigned short*)p; p += al((size_t)64 * 128 * 2);
    unsigned short* w2b  = (unsigned short*)p; p += al((size_t)128 * 64 * 2);
    unsigned short* embb = (unsigned short*)p; p += al((size_t)EMB * 2);
    unsigned short* t1   = (unsigned short*)p; p += al((size_t)N * 128 * 2); // also h1
    float*          agg1 = (float*)p;          p += al((size_t)N * 128 * 4);
    // after conv1 is consumed, reuse agg1 region:
    float*          agg2 = agg1;                              // N*64 floats
    unsigned short* t2   = (unsigned short*)(agg1 + (size_t)N * 64);

    const int B = 256;
    auto cdiv = [](long long a, long long b) { return (int)((a + b - 1) / b); };

    // weights / embeddings -> bf16
    f32_to_bf16_kernel<<<cdiv(64 * 128, B), B, 0, stream>>>(W1, w1b, 64 * 128);
    f32_to_bf16_kernel<<<cdiv(128 * 64, B), B, 0, stream>>>(W2, w2b, 128 * 64);
    f32_to_bf16_kernel<<<cdiv(EMB, B), B, 0, stream>>>(embt, embb, EMB);

    // degree -> dinv
    zero_kernel<<<cdiv(N, B), B, 0, stream>>>(deg, (size_t)N);
    zero_kernel<<<cdiv((long long)N * 128, B), B, 0, stream>>>(agg1, (size_t)N * 128);
    degree_kernel<<<cdiv(E, B), B, 0, stream>>>(col, eattr, deg, E);
    dinv_kernel<<<cdiv(N, B), B, 0, stream>>>(deg, N);

    // conv1: t1 = embed[x_ids] @ W1  (WMMA)
    gemm1_kernel<<<cdiv(N, 16 * 8), B, 0, stream>>>(x_ids, embb, w1b, t1, N, 8);
    scatter_kernel<128><<<cdiv((long long)(E + N) * 32, B), B, 0, stream>>>(
        row, col, eattr, deg, t1, agg1, E, N);
    bias_relu_kernel<<<cdiv((long long)N * 128, B), B, 0, stream>>>(agg1, b1, t1, N);

    // conv2: t2 = h1 @ W2  (WMMA)
    zero_kernel<<<cdiv((long long)N * 64, B), B, 0, stream>>>(agg2, (size_t)N * 64);
    gemm2_kernel<<<cdiv(N, 32 * 8), B, 0, stream>>>(t1, w2b, t2, N, 8);
    scatter_kernel<64><<<cdiv((long long)(E + N) * 32, B), B, 0, stream>>>(
        row, col, eattr, deg, t2, agg2, E, N);

    // global max pool (+b2), decode in place
    pool_init_kernel<<<cdiv(G * 64, B), B, 0, stream>>>((unsigned*)d_out, G * 64);
    pool_max_kernel<<<cdiv((long long)N * 64, B), B, 0, stream>>>(
        batch, agg2, b2, (unsigned*)d_out, N);
    pool_fin_kernel<<<cdiv(G * 64, B), B, 0, stream>>>((unsigned*)d_out, G * 64);
}
